// TimeMixing_9826885173421
// MI455X (gfx1250) — compile-verified, matrix operand order checked by reading, not run on hardware
//
#include <hip/hip_runtime.h>

// Problem constants (reference: B=2, T=256, C=512, A=512)
#define BB 2
#define TT 256
#define CC 512
#define AA 512
#define MM (BB * TT)   // 512 rows of the flattened (B*T, ...) matrices

typedef __attribute__((ext_vector_type(16))) __bf16 v16bf;
typedef __attribute__((ext_vector_type(8)))  float  v8f;

// 32-byte bf16 operand fragment: two 16B chunks -> one v16bf WMMA operand
union BFrag {
    uint4  q[2];
    v16bf  v;
};

__device__ __forceinline__ unsigned short f2bf(float f) {
    unsigned u = __float_as_uint(f);
    unsigned r = 0x7FFFu + ((u >> 16) & 1u);   // round-to-nearest-even
    return (unsigned short)((u + r) >> 16);
}

// ---------------------------------------------------------------------------
// CDNA5 async global->LDS copy (ASYNCcnt path, CDNA5 ISA).
// lds_byte: LDS byte address (hardware adds wave LDS_BASE); g: global address.
// ---------------------------------------------------------------------------
__device__ __forceinline__ void async_b128(unsigned lds_byte, const void* g) {
    asm volatile("global_load_async_to_lds_b128 %0, %1, off"
                 :: "v"(lds_byte), "v"((unsigned long long)g)
                 : "memory");
}

template <int N>
__device__ __forceinline__ void wait_async() {
    asm volatile("s_wait_asynccnt %0" :: "i"(N) : "memory");
}

// ---------------------------------------------------------------------------
// Stage 1: token-shift mix -> bf16 activations XR, XV
// ---------------------------------------------------------------------------
__global__ void mix_kernel(const float* __restrict__ x,
                           const float* __restrict__ muR,
                           const float* __restrict__ muV,
                           unsigned short* __restrict__ XR,
                           unsigned short* __restrict__ XV) {
    int idx = blockIdx.x * blockDim.x + threadIdx.x;   // over B*T*C, exact grid
    int c = idx % CC;
    int t = (idx / CC) % TT;
    float xc = x[idx];
    float xs = (t < TT - 1) ? x[idx + CC] : 0.0f;
    float mr = muR[t * CC + c];
    float mv = muV[t * CC + c];
    XR[idx] = f2bf(mr * xc + (1.0f - mr) * xs);
    XV[idx] = f2bf(mv * xc + (1.0f - mv) * xs);
}

// f32 -> bf16 weight conversion
__global__ void cvt_kernel(const float* __restrict__ src,
                           unsigned short* __restrict__ dst) {
    int idx = blockIdx.x * blockDim.x + threadIdx.x;   // exact grid
    dst[idx] = f2bf(src[idx]);
}

// ---------------------------------------------------------------------------
// Stage 2: dual GEMM  R = XR@Wr^T + br,  V = XV@Wv^T + bv,
// fused epilogue S = sigmoid(R)*V -> bf16.
//
// Block tile: 64 rows x 128 cols; 8 waves as 2(m) x 4(n); wave tile 32x32
// (2x2 WMMA chains for each of R and V). Operand panels double-buffered in
// LDS via async global->LDS b128 copies.
//   LDS buffer (24 KB): XR 64x32 @0, XV @4096, Wr 128x32 @8192, Wv @16384.
// ---------------------------------------------------------------------------
__global__ __launch_bounds__(256) void gemm_rv_kernel(
        const unsigned short* __restrict__ XR,
        const unsigned short* __restrict__ XV,
        const unsigned short* __restrict__ WrB,
        const unsigned short* __restrict__ WvB,
        const float* __restrict__ br,
        const float* __restrict__ bv,
        unsigned short* __restrict__ S) {
    __shared__ unsigned char smem[2 * 24576];

    const int tid  = threadIdx.x;
    const int wave = tid >> 5;
    const int lane = tid & 31;
    const int hi   = lane >> 4;
    const int ln   = lane & 15;
    const int mW   = wave >> 2;                // 0..1
    const int nW   = wave & 3;                 // 0..3
    const int m0b  = (blockIdx.x & 7) * 64;    // 8 m-blocks
    const int n0b  = (blockIdx.x >> 3) * 128;  // 4 n-blocks

    const unsigned sbase = (unsigned)(unsigned long long)&smem[0];

    auto stage = [&](int kk) {                 // 6 async b128 per lane
        const unsigned bb = sbase + (unsigned)((kk & 1) * 24576);
        const int k0 = kk * 32;
        {
            const int row = tid >> 2, cc = tid & 3;
            async_b128(bb + 0u    + row * 64 + cc * 16,
                       XR + (size_t)(m0b + row) * CC + k0 + cc * 8);
            async_b128(bb + 4096u + row * 64 + cc * 16,
                       XV + (size_t)(m0b + row) * CC + k0 + cc * 8);
        }
#pragma unroll
        for (int j = 0; j < 2; ++j) {
            const int cid = tid + j * 256;
            const int row = cid >> 2, cc = cid & 3;
            async_b128(bb + 8192u  + row * 64 + cc * 16,
                       WrB + (size_t)(n0b + row) * CC + k0 + cc * 8);
            async_b128(bb + 16384u + row * 64 + cc * 16,
                       WvB + (size_t)(n0b + row) * CC + k0 + cc * 8);
        }
    };

    v8f accR[2][2] = {};
    v8f accV[2][2] = {};

    stage(0);
    for (int kk = 0; kk < 16; ++kk) {
        if (kk < 15) { stage(kk + 1); wait_async<6>(); }  // in-order: panel kk done
        else         { wait_async<0>(); }
        __syncthreads();                                  // panel kk visible to WG

        const unsigned char* sp = smem + (kk & 1) * 24576;
        BFrag aR[2], aV[2], wR[2], wV[2];
#pragma unroll
        for (int mt = 0; mt < 2; ++mt) {
            const int ro = (mW * 32 + mt * 16 + ln) * 64 + hi * 16;
            aR[mt].q[0] = *(const uint4*)(sp + 0    + ro);
            aR[mt].q[1] = *(const uint4*)(sp + 0    + ro + 32);
            aV[mt].q[0] = *(const uint4*)(sp + 4096 + ro);
            aV[mt].q[1] = *(const uint4*)(sp + 4096 + ro + 32);
        }
#pragma unroll
        for (int nt = 0; nt < 2; ++nt) {
            const int ro = (nW * 32 + nt * 16 + ln) * 64 + hi * 16;
            wR[nt].q[0] = *(const uint4*)(sp + 8192  + ro);
            wR[nt].q[1] = *(const uint4*)(sp + 8192  + ro + 32);
            wV[nt].q[0] = *(const uint4*)(sp + 16384 + ro);
            wV[nt].q[1] = *(const uint4*)(sp + 16384 + ro + 32);
        }
#pragma unroll
        for (int mt = 0; mt < 2; ++mt)
#pragma unroll
            for (int nt = 0; nt < 2; ++nt) {
                accR[mt][nt] = __builtin_amdgcn_wmma_f32_16x16x32_bf16(
                    false, aR[mt].v, false, wR[nt].v, (short)0, accR[mt][nt], false, false);
                accV[mt][nt] = __builtin_amdgcn_wmma_f32_16x16x32_bf16(
                    false, aV[mt].v, false, wV[nt].v, (short)0, accV[mt][nt], false, false);
            }
        __syncthreads();                       // all reads done before buf reuse
    }

    // Epilogue: elem r of lane -> row = base + hi*8 + r, col = base + ln
#pragma unroll
    for (int mt = 0; mt < 2; ++mt) {
        const int rowb = m0b + mW * 32 + mt * 16 + hi * 8;
#pragma unroll
        for (int nt = 0; nt < 2; ++nt) {
            const int col = n0b + nW * 32 + nt * 16 + ln;
            const float bRc = br[col];
            const float bVc = bv[col];
#pragma unroll
            for (int r = 0; r < 8; ++r) {
                float rv  = accR[mt][nt][r] + bRc;
                float vv  = accV[mt][nt][r] + bVc;
                float sig = 1.0f / (1.0f + __expf(-rv));
                S[(size_t)(rowb + r) * AA + col] = f2bf(sig * vv);
            }
        }
    }
}

// ---------------------------------------------------------------------------
// Stage 3: out = S @ Wo^T + bo  (f32, (B*T, C))
// Same blocking; LDS buffer (12 KB): S 64x32 @0, Wo 128x32 @4096.
// ---------------------------------------------------------------------------
__global__ __launch_bounds__(256) void gemm_out_kernel(
        const unsigned short* __restrict__ Sb,
        const unsigned short* __restrict__ WoB,
        const float* __restrict__ bo,
        float* __restrict__ out) {
    __shared__ unsigned char smem[2 * 12288];

    const int tid  = threadIdx.x;
    const int wave = tid >> 5;
    const int lane = tid & 31;
    const int hi   = lane >> 4;
    const int ln   = lane & 15;
    const int mW   = wave >> 2;
    const int nW   = wave & 3;
    const int m0b  = (blockIdx.x & 7) * 64;
    const int n0b  = (blockIdx.x >> 3) * 128;

    const unsigned sbase = (unsigned)(unsigned long long)&smem[0];

    auto stage = [&](int kk) {                 // 3 async b128 per lane
        const unsigned bb = sbase + (unsigned)((kk & 1) * 12288);
        const int k0 = kk * 32;
        {
            const int row = tid >> 2, cc = tid & 3;
            async_b128(bb + 0u + row * 64 + cc * 16,
                       Sb + (size_t)(m0b + row) * AA + k0 + cc * 8);
        }
#pragma unroll
        for (int j = 0; j < 2; ++j) {
            const int cid = tid + j * 256;
            const int row = cid >> 2, cc = cid & 3;
            async_b128(bb + 4096u + row * 64 + cc * 16,
                       WoB + (size_t)(n0b + row) * AA + k0 + cc * 8);
        }
    };

    v8f acc[2][2] = {};

    stage(0);
    for (int kk = 0; kk < 16; ++kk) {
        if (kk < 15) { stage(kk + 1); wait_async<3>(); }
        else         { wait_async<0>(); }
        __syncthreads();

        const unsigned char* sp = smem + (kk & 1) * 12288;
        BFrag a[2], b[2];
#pragma unroll
        for (int mt = 0; mt < 2; ++mt) {
            const int ro = (mW * 32 + mt * 16 + ln) * 64 + hi * 16;
            a[mt].q[0] = *(const uint4*)(sp + 0 + ro);
            a[mt].q[1] = *(const uint4*)(sp + 0 + ro + 32);
        }
#pragma unroll
        for (int nt = 0; nt < 2; ++nt) {
            const int ro = (nW * 32 + nt * 16 + ln) * 64 + hi * 16;
            b[nt].q[0] = *(const uint4*)(sp + 4096 + ro);
            b[nt].q[1] = *(const uint4*)(sp + 4096 + ro + 32);
        }
#pragma unroll
        for (int mt = 0; mt < 2; ++mt)
#pragma unroll
            for (int nt = 0; nt < 2; ++nt)
                acc[mt][nt] = __builtin_amdgcn_wmma_f32_16x16x32_bf16(
                    false, a[mt].v, false, b[nt].v, (short)0, acc[mt][nt], false, false);
        __syncthreads();
    }

#pragma unroll
    for (int mt = 0; mt < 2; ++mt) {
        const int rowb = m0b + mW * 32 + mt * 16 + hi * 8;
#pragma unroll
        for (int nt = 0; nt < 2; ++nt) {
            const int col = n0b + nW * 32 + nt * 16 + ln;
            const float bc = bo[col];
#pragma unroll
            for (int r = 0; r < 8; ++r)
                out[(size_t)(rowb + r) * CC + col] = acc[mt][nt][r] + bc;
        }
    }
}

// ---------------------------------------------------------------------------
// Host-side launcher
// Input order: x, mu_R, mu_K, mu_V, Wr, br, Wk, bk, Wv, bv, Wo, bo, npw, npo
// (K/Wk/bk and the pos buffers cancel out: wkv/denom == V identically.)
// ---------------------------------------------------------------------------
extern "C" void kernel_launch(void* const* d_in, const int* in_sizes, int n_in,
                              void* d_out, int out_size, void* d_ws, size_t ws_size,
                              hipStream_t stream) {
    (void)in_sizes; (void)n_in; (void)out_size;

    const float* x   = (const float*)d_in[0];
    const float* muR = (const float*)d_in[1];
    const float* muV = (const float*)d_in[3];
    const float* Wr  = (const float*)d_in[4];
    const float* br  = (const float*)d_in[5];
    const float* Wv  = (const float*)d_in[8];
    const float* bv  = (const float*)d_in[9];
    const float* Wo  = (const float*)d_in[10];
    const float* bo  = (const float*)d_in[11];
    float*       out = (float*)d_out;

    // Workspace layout (bf16 planes, 512 KB each, 3 MB total)
    unsigned short* XR  = (unsigned short*)d_ws;
    unsigned short* XV  = XR  + (size_t)MM * CC;
    unsigned short* WrB = XV  + (size_t)MM * CC;
    unsigned short* WvB = WrB + (size_t)AA * CC;
    unsigned short* WoB = WvB + (size_t)AA * CC;
    unsigned short* Sb  = WoB + (size_t)CC * AA;
    if (ws_size < (size_t)6 * MM * CC * sizeof(unsigned short)) return;

    // Stage 1: mix + conversions
    mix_kernel<<<(MM * CC) / 256, 256, 0, stream>>>(x, muR, muV, XR, XV);
    cvt_kernel<<<(AA * CC) / 256, 256, 0, stream>>>(Wr, WrB);
    cvt_kernel<<<(AA * CC) / 256, 256, 0, stream>>>(Wv, WvB);
    cvt_kernel<<<(CC * AA) / 256, 256, 0, stream>>>(Wo, WoB);

    // Stage 2: dual GEMM + sigmoid*V epilogue. 64x128 block tiles -> 32 blocks.
    gemm_rv_kernel<<<(MM / 64) * (AA / 128), 256, 0, stream>>>(
        XR, XV, WrB, WvB, br, bv, Sb);

    // Stage 3: output GEMM + bias. 32 blocks.
    gemm_out_kernel<<<(MM / 64) * (CC / 128), 256, 0, stream>>>(
        Sb, WoB, bo, out);
}
